// customAttention_48515950576474
// MI455X (gfx1250) — compile-verified, matrix operand order checked by reading, not run on hardware
//
#include <hip/hip_runtime.h>
#include <hip/hip_fp16.h>

typedef __attribute__((ext_vector_type(16))) _Float16 v16h;
typedef __attribute__((ext_vector_type(8)))  float    v8f;
typedef __attribute__((ext_vector_type(4)))  float    v4f;
typedef __attribute__((ext_vector_type(2)))  float    v2f;

namespace {
constexpr int B_TOT  = 8192;
constexpr int MSTEP  = 200;
constexpr int NFEAT  = 64;
constexpr int ATT    = 64;
constexpr int MTILES = 13;     // ceil(200/16); rows 200..207 are padding
constexpr int KEY_LD = 66;     // even stride: 8B-aligned rows, banks (2m+a)%64 conflict-free
constexpr int WK_LD  = 65;     // f16 LDS row stride for staged Wk
}

__device__ __forceinline__ float fast_tanh(float v) {
#if __has_builtin(__builtin_amdgcn_tanhf)
    return __builtin_amdgcn_tanhf(v);    // hardware v_tanh_f32 (TRANS op on CDNA5)
#else
    return tanhf(v);
#endif
}

__launch_bounds__(256)
__global__ void attn_additive_kernel(const float* __restrict__ x,
                                     const float* __restrict__ mem,
                                     const float* __restrict__ Wq,
                                     const float* __restrict__ Wk,
                                     const float* __restrict__ bias,
                                     const float* __restrict__ Ws,
                                     float* __restrict__ out)
{
    __shared__ __align__(16) float key[MTILES * 16 * KEY_LD];   // 208 x 66 f32 = 54,912 B
    // aux overlays: [phase 0-1] Wk staged as f16 (64x65 = 8,320 B)
    //               [phase 4-5] wts (200 f32 @ 0) + cpart (8x64 f32 @ 1024)
    __shared__ __align__(16) unsigned char aux[8448];
    __shared__ __align__(16) float qb[ATT];
    __shared__ __align__(16) float wsh[ATT];
    __shared__ float redmax[8];
    __shared__ float redsum[8];

    _Float16* wkh   = reinterpret_cast<_Float16*>(aux);
    float*    wts   = reinterpret_cast<float*>(aux);
    float*    cpart = reinterpret_cast<float*>(aux + 1024);     // [8][64]

    const int b    = blockIdx.x;
    const int tid  = threadIdx.x;
    const int lane = tid & 31;
    const int wave = tid >> 5;

    // ---- Phase 0: stage Wk (f32->f16) into LDS; q = x@Wq + bias; stage Ws ----
    for (int i = tid; i < NFEAT * ATT; i += 256) {
        const int k = i >> 6, n = i & 63;
        wkh[k * WK_LD + n] = (_Float16)Wk[i];
    }
    if (tid < ATT) {
        float acc = bias[tid];
        const float* xb = x + (size_t)b * NFEAT;
        #pragma unroll 8
        for (int f = 0; f < NFEAT; ++f) acc = fmaf(xb[f], Wq[f * ATT + tid], acc);
        qb[tid] = acc;
    } else if (tid < 2 * ATT) {
        wsh[tid - ATT] = Ws[tid - ATT];
    }
    __syncthreads();

    // ---- Phase 1: build this wave's B fragments (Wk N-tile), keep in registers ----
    // B 32x16 f16 layout: lane (0..15)=N col; lanes 16..31 same cols, upper K octet.
    const int nt    = wave & 3;          // waves 0-3 and 4-7 each cover nt = 0..3
    const int nloc  = lane & 15;
    const int khalf = (lane >> 4) << 3;  // 0 or 8
    v16h bfrag[2];
    #pragma unroll
    for (int kk = 0; kk < 2; ++kk) {
        v16h f;
        #pragma unroll
        for (int h = 0; h < 16; ++h) {
            const int k = kk * 32 + ((h < 8) ? (khalf + h) : (16 + khalf + (h - 8)));
            f[h] = wkh[k * WK_LD + nt * 16 + nloc];
        }
        bfrag[kk] = f;
    }

    // ---- Phase 2: key = mem[b] @ Wk via v_wmma_f32_16x16x32_f16, K = 64 = 2x32 ----
    const float* memb = mem + (size_t)b * MSTEP * NFEAT;
    for (int mt = (wave >> 2); mt < MTILES; mt += 2) {
        int row = mt * 16 + nloc;
        if (row > MSTEP - 1) row = MSTEP - 1;        // clamp pad rows (never read back)
        const float* arow = memb + row * NFEAT;
        v8f c = {};
        #pragma unroll
        for (int kk = 0; kk < 2; ++kk) {
            const int kb = kk * 32 + khalf;
            const v4f f0 = __builtin_nontemporal_load((const v4f*)(arow + kb));
            const v4f f1 = __builtin_nontemporal_load((const v4f*)(arow + kb + 4));
            const v4f f2 = __builtin_nontemporal_load((const v4f*)(arow + kb + 16));
            const v4f f3 = __builtin_nontemporal_load((const v4f*)(arow + kb + 20));
            v16h a;
            #pragma unroll
            for (int j = 0; j < 4; ++j) {
                a[j]      = (_Float16)f0[j];
                a[4 + j]  = (_Float16)f1[j];
                a[8 + j]  = (_Float16)f2[j];
                a[12 + j] = (_Float16)f3[j];
            }
            c = __builtin_amdgcn_wmma_f32_16x16x32_f16(false, a, false, bfrag[kk],
                                                       (short)0, c, false, false);
        }
        const int cn = nt * 16 + nloc;
        const int m0 = mt * 16 + ((lane >> 4) << 3);
        #pragma unroll
        for (int r = 0; r < 8; ++r)
            key[(m0 + r) * KEY_LD + cn] = c[r];
    }
    __syncthreads();   // also orders last wkh reads before wts/cpart overlay writes

    // ---- Phase 3: scores[m] = sum_a tanh(q[a]+key[m][a]) * Ws[a] ----
    float score = 0.f;
    if (tid < MSTEP) {
        const v2f* krow = (const v2f*)(key + tid * KEY_LD);   // 8B aligned (KEY_LD even)
        const v2f* q2   = (const v2f*)qb;
        const v2f* w2   = (const v2f*)wsh;
        #pragma unroll 8
        for (int j = 0; j < ATT / 2; ++j) {
            const v2f k = krow[j];
            const v2f q = q2[j];
            const v2f w = w2[j];
            score = fmaf(fast_tanh(q[0] + k[0]), w[0], score);
            score = fmaf(fast_tanh(q[1] + k[1]), w[1], score);
        }
    }

    // ---- Phase 4: softmax over m (wave32 shuffle + cross-wave LDS combine) ----
    float v = (tid < MSTEP) ? score : -3.0e38f;
    #pragma unroll
    for (int off = 16; off > 0; off >>= 1) v = fmaxf(v, __shfl_xor(v, off, 32));
    if (lane == 0) redmax[wave] = v;
    __syncthreads();
    float gmax = redmax[0];
    #pragma unroll
    for (int w = 1; w < 8; ++w) gmax = fmaxf(gmax, redmax[w]);

    const float e = (tid < MSTEP) ? __expf(score - gmax) : 0.f;
    float s = e;
    #pragma unroll
    for (int off = 16; off > 0; off >>= 1) s += __shfl_xor(s, off, 32);
    if (lane == 0) redsum[wave] = s;
    __syncthreads();
    float gsum = 0.f;
    #pragma unroll
    for (int w = 0; w < 8; ++w) gsum += redsum[w];

    if (tid < MSTEP) wts[tid] = e / gsum;
    __syncthreads();

    // ---- Phase 5: context[a] = sum_m wts[m] * key[m][a]  (v2f per thread) ----
    {
        const int p = tid & 31;            // a-pair: covers a = 2p, 2p+1
        const int g = tid >> 5;            // 8 groups x 25 rows
        const v2f* keyv = (const v2f*)key; // row stride KEY_LD/2 = 33 v2f
        v2f acc = {0.f, 0.f};
        for (int m = g * 25; m < g * 25 + 25; ++m) {
            const v2f k = keyv[m * (KEY_LD / 2) + p];
            const float w = wts[m];
            acc[0] = fmaf(w, k[0], acc[0]);
            acc[1] = fmaf(w, k[1], acc[1]);
        }
        cpart[g * ATT + 2 * p]     = acc[0];
        cpart[g * ATT + 2 * p + 1] = acc[1];
    }
    __syncthreads();
    if (tid < ATT) {
        float acc = 0.f;
        #pragma unroll
        for (int g = 0; g < 8; ++g) acc += cpart[g * ATT + tid];
        out[(size_t)b * ATT + tid] = acc;
    }
}

extern "C" void kernel_launch(void* const* d_in, const int* in_sizes, int n_in,
                              void* d_out, int out_size, void* d_ws, size_t ws_size,
                              hipStream_t stream) {
    const float* x    = (const float*)d_in[0];
    const float* mem  = (const float*)d_in[1];
    const float* Wq   = (const float*)d_in[2];
    const float* Wk   = (const float*)d_in[3];
    const float* bias = (const float*)d_in[4];
    const float* Ws   = (const float*)d_in[5];
    float* out        = (float*)d_out;

    attn_additive_kernel<<<dim3(B_TOT), dim3(256), 0, stream>>>(
        x, mem, Wq, Wk, bias, Ws, out);
}
